// MultiScaleAttentionLayer_15479062135318
// MI455X (gfx1250) — matrix operand order repaired and checked by reading, not compiled, not run on hardware
//
#include <hip/hip_runtime.h>
#include <cmath>
#include <cstdint>

// ---------------------------------------------------------------------------
// MI455X (gfx1250, wave32) implementation of the GAT + Performer + FFN layer.
// Dense GEMMs run on v_wmma_f32_16x16x32_f16 (f16 operands, f32 accumulate)
// with 16x64 register-blocked C-tiles per wave (A fragment reused across 4
// WMMAs). KV reduction uses 32x32 tiles with split-K f32 atomic combine.
// Sparse segment-softmax uses float-ordered uint atomicMax + atomicAdd.
// Everything is stream-ordered / graph-safe.
// ---------------------------------------------------------------------------

typedef _Float16 v16h __attribute__((ext_vector_type(16)));
typedef _Float16 v8h  __attribute__((ext_vector_type(8)));
typedef float    v8f  __attribute__((ext_vector_type(8)));

#define DEV __device__ __forceinline__

constexpr int NN   = 50000;   // nodes
constexpr int EE   = 400000;  // edges
constexpr int DD   = 128;     // feature dim
constexpr int HD   = 512;     // H * D
constexpr int NPAD = 50016;   // NN padded to multiple of 32 (KV reduction K-dim)

// ---------------- small helpers ----------------
DEV unsigned f32_enc(float f) {            // order-preserving encode for atomicMax
  unsigned u = __float_as_uint(f);
  return (u & 0x80000000u) ? ~u : (u | 0x80000000u);
}
DEV float f32_dec(unsigned u) {
  return (u & 0x80000000u) ? __uint_as_float(u ^ 0x80000000u) : __uint_as_float(~u);
}
DEV float gelu_exact(float x) { return 0.5f * x * (1.0f + erff(x * 0.70710678118654752f)); }

// ---------------- utility kernels ----------------
__global__ __launch_bounds__(256) void k_zero_u32(unsigned* p, size_t n) {
  size_t i = (size_t)blockIdx.x * 256 + threadIdx.x;
  size_t stride = (size_t)gridDim.x * 256;
  for (; i < n; i += stride) p[i] = 0u;
}
__global__ void k_init_amax(unsigned* p) { *p = f32_enc(-3.0e38f); }

__global__ __launch_bounds__(256) void k_f16cvt(_Float16* dst, const float* src, int n) {
  int i = blockIdx.x * 256 + threadIdx.x;
  int stride = gridDim.x * 256;
  for (; i < n; i += stride) dst[i] = (_Float16)src[i];
}

// ---------------- generic WMMA GEMM: C[r][c] = act( sum_k A[r][k]*B[c][k] ) ----
// A: [R, Kd] f16 row-major (K contiguous). B: [C, Kd] f16 row-major (K contig).
// One wave per 16x64 C-tile (4 accumulators); K stepped by 32.
struct EpiP {
  float*          of;     // f32 output (x_state / d_out)
  _Float16*       oh;     // f16 output
  const float*    bias;
  const float*    resid;
  const float*    zbuf;
  int             Cfull;  // logical C columns
  int             ldT;    // leading dim for transposed stores (NPAD)
  int             Nnodes; // N
  float           dn;     // performer data normalizer
};

enum {
  M_STORE16 = 0,  // oh[row*Cfull+col] = v
  M_STOREVT = 1,  // oh[col*ldT + row] = v                   (V^T staging)
  M_QP      = 2,  // Qph[(h*N+n)*128 + col] = relu(dn*v)+eps (head-major)
  M_KP      = 3,  // KpT[(h*128+col)*ldT + n] = relu(dn*v)+eps
  M_OUTEIN  = 4,  // AOh[n*512 + h*128 + col] = v * z[row]; B offset per head
  M_RESADD  = 5,  // of[row*Cfull+col] += v + bias[col]      (x_state += attn)
  M_GELU16  = 6,  // oh[row*Cfull+col] = gelu(v + bias[col])
  M_FINAL   = 7   // of[idx] = resid[idx] + v + bias[col]    (d_out)
};

template <int MODE>
__global__ __launch_bounds__(256) void k_gemm(const _Float16* __restrict__ A,
                                              const _Float16* __restrict__ B,
                                              int Kd, int cblocks, int totalTiles,
                                              EpiP P) {
  const int wave = threadIdx.x >> 5;
  const int lane = threadIdx.x & 31;
  const int tile = blockIdx.x * 8 + wave;
  if (tile >= totalTiles) return;                 // wave-uniform (EXEC stays all-1)
  const int rt = tile / cblocks;                  // 16-row tile index
  const int cb = tile % cblocks;                  // 64-column block index
  const int hf  = lane >> 4;                      // half-wave select
  const int l16 = lane & 15;

  const _Float16* Bp = B;
  if (MODE == M_OUTEIN) {                         // per-head B (KV^T), tiles never straddle heads
    int h = (rt * 16) / P.Nnodes;
    Bp += (size_t)h * (128 * 128);
  }
  const _Float16* Arow = A + (size_t)(rt * 16 + l16) * Kd;
  const _Float16* Br[4];
#pragma unroll
  for (int j = 0; j < 4; ++j)
    Br[j] = Bp + (size_t)(cb * 64 + j * 16 + l16) * Kd;

  v8f acc[4];
  const v8f vz = {0.f, 0.f, 0.f, 0.f, 0.f, 0.f, 0.f, 0.f};
#pragma unroll
  for (int j = 0; j < 4; ++j) acc[j] = vz;

  for (int k = 0; k < Kd; k += 32) {
    // A fragment: lane(half,m): K = 16*hf + 0..15 (contiguous 32B)
    v16h a = *(const v16h*)(Arow + k + 16 * hf);
    // B fragments: lane(half,n): K = 16*hf + 0..15 (contiguous 32B); A reused 4x
#pragma unroll
    for (int j = 0; j < 4; ++j) {
      v16h b = *(const v16h*)(Br[j] + k + 16 * hf);
      acc[j] = __builtin_amdgcn_wmma_f32_16x16x32_f16(false, a, false, b, (short)0,
                                                      acc[j], false, false);
    }
  }

  // epilogue: acc[j][r] -> C[rt*16 + r + 8*hf][cb*64 + j*16 + l16]
#pragma unroll
  for (int j = 0; j < 4; ++j) {
#pragma unroll
    for (int r = 0; r < 8; ++r) {
      const int row = rt * 16 + r + 8 * hf;
      const int col = cb * 64 + j * 16 + l16;
      float v = acc[j][r];
      if (MODE == M_STORE16) {
        P.oh[(size_t)row * P.Cfull + col] = (_Float16)v;
      } else if (MODE == M_STOREVT) {
        P.oh[(size_t)col * P.ldT + row] = (_Float16)v;
      } else if (MODE == M_QP) {
        int h = row & 3, n = row >> 2;
        float q = fmaxf(P.dn * v, 0.f) + 1.0e-3f;
        P.oh[((size_t)h * P.Nnodes + n) * 128 + col] = (_Float16)q;
      } else if (MODE == M_KP) {
        int h = row & 3, n = row >> 2;
        float q = fmaxf(P.dn * v, 0.f) + 1.0e-3f;
        P.oh[((size_t)(h * 128 + col)) * P.ldT + n] = (_Float16)q;
      } else if (MODE == M_OUTEIN) {
        int h = row / P.Nnodes, n = row % P.Nnodes;
        v *= P.zbuf[row];
        P.oh[(size_t)n * 512 + h * 128 + col] = (_Float16)v;
      } else if (MODE == M_RESADD) {
        P.of[(size_t)row * P.Cfull + col] += v + P.bias[col];
      } else if (MODE == M_GELU16) {
        P.oh[(size_t)row * P.Cfull + col] = (_Float16)gelu_exact(v + P.bias[col]);
      } else if (MODE == M_FINAL) {
        size_t idx = (size_t)row * P.Cfull + col;
        P.of[idx] = P.resid[idx] + v + P.bias[col];
      }
    }
  }
}

// ---------------- KV = Kp^T @ V per head (split-K WMMA, f32 atomic combine) ----
// One wave per 32x32 C-tile (2x2 register blocking), K split across waves.
__global__ __launch_bounds__(256) void k_kv(const _Float16* __restrict__ KpT,
                                            const _Float16* __restrict__ VhT,
                                            float* __restrict__ KV,
                                            int ksplit, int stepsTotal) {
  const int wave = threadIdx.x >> 5;
  const int lane = threadIdx.x & 31;
  int flat = blockIdx.x * 8 + wave;
  const int total = 4 * 4 * 4 * ksplit;
  if (flat >= total) return;
  const int chunk = flat % ksplit; flat /= ksplit;
  const int dt = flat % 4;         flat /= 4;
  const int mt = flat % 4;
  const int h  = flat / 4;
  const int per = (stepsTotal + ksplit - 1) / ksplit;
  const int k0 = chunk * per * 32;
  const int ke = ((chunk + 1) * per < stepsTotal ? (chunk + 1) * per : stepsTotal) * 32;
  if (k0 >= ke) return;                            // wave-uniform

  const int hf  = lane >> 4;
  const int l16 = lane & 15;
  const _Float16* Ar[2];
  const _Float16* Brw[2];
#pragma unroll
  for (int i = 0; i < 2; ++i) {
    Ar[i]  = KpT + (size_t)(h * 128 + mt * 32 + i * 16 + l16) * NPAD;
    Brw[i] = VhT + (size_t)(h * 128 + dt * 32 + i * 16 + l16) * NPAD;
  }

  v8f acc[2][2];
  const v8f vz = {0.f, 0.f, 0.f, 0.f, 0.f, 0.f, 0.f, 0.f};
#pragma unroll
  for (int i = 0; i < 2; ++i)
#pragma unroll
    for (int j = 0; j < 2; ++j) acc[i][j] = vz;

  for (int k = k0; k < ke; k += 32) {
    v16h a[2], b[2];
#pragma unroll
    for (int i = 0; i < 2; ++i) {
      a[i] = *(const v16h*)(Ar[i] + k + 16 * hf);
      b[i] = *(const v16h*)(Brw[i] + k + 16 * hf);
    }
#pragma unroll
    for (int i = 0; i < 2; ++i)
#pragma unroll
      for (int j = 0; j < 2; ++j)
        acc[i][j] = __builtin_amdgcn_wmma_f32_16x16x32_f16(false, a[i], false, b[j],
                                                           (short)0, acc[i][j],
                                                           false, false);
  }
#pragma unroll
  for (int i = 0; i < 2; ++i)
#pragma unroll
    for (int j = 0; j < 2; ++j)
#pragma unroll
      for (int r = 0; r < 8; ++r) {
        int m = mt * 32 + i * 16 + r + 8 * hf;
        int d = dt * 32 + j * 16 + l16;
        atomicAdd(&KV[(size_t)h * 16384 + (size_t)m * 128 + d], acc[i][j][r]);
      }
}

// ---------------- GAT attention-logit dots: al/ar per (node, head) -----------
__global__ __launch_bounds__(256) void k_alar(const _Float16* __restrict__ xph,
                                              const float* __restrict__ attn_l,
                                              const float* __restrict__ attn_r,
                                              float* __restrict__ al,
                                              float* __restrict__ ar) {
  int t = blockIdx.x * 256 + threadIdx.x;       // t = n*4 + h
  if (t >= NN * 4) return;
  int n = t >> 2, h = t & 3;
  const _Float16* xr = xph + (size_t)n * HD + h * DD;
  float sl = 0.f, sr = 0.f;
  for (int d = 0; d < DD; ++d) {
    float v = (float)xr[d];
    sl += v * attn_l[h * DD + d];
    sr += v * attn_r[h * DD + d];
  }
  al[t] = sl; ar[t] = sr;
}

// ---------------- edge pass 1: leaky_relu logits + global max ----------------
__global__ __launch_bounds__(256) void k_edge1(const int* __restrict__ ei,
                                               const float* __restrict__ al,
                                               const float* __restrict__ ar,
                                               float* __restrict__ ae,
                                               unsigned* __restrict__ amax) {
  int e = blockIdx.x * 256 + threadIdx.x;
  unsigned enc = 0u;
  if (e < EE) {
    int r = ei[e], c = ei[EE + e];
    float mx = -3.0e38f;
#pragma unroll
    for (int h = 0; h < 4; ++h) {
      float v = al[r * 4 + h] + ar[c * 4 + h];
      float a = v > 0.f ? v : 0.2f * v;
      ae[(size_t)e * 4 + h] = a;
      mx = fmaxf(mx, a);
    }
    enc = f32_enc(mx);
  }
  for (int off = 16; off > 0; off >>= 1) {
    unsigned o = (unsigned)__shfl_xor((int)enc, off, 32);
    enc = enc > o ? enc : o;
  }
  if ((threadIdx.x & 31) == 0) atomicMax(amax, enc);
}

// ---------------- edge pass 2: w = exp(a - amax); segment max ----------------
__global__ __launch_bounds__(256) void k_edge2(const int* __restrict__ ei,
                                               float* __restrict__ ae,
                                               const unsigned* __restrict__ amax,
                                               unsigned* __restrict__ mseg) {
  int e = blockIdx.x * 256 + threadIdx.x;
  if (e >= EE) return;
  float A = f32_dec(*amax);
  int r = ei[e];
#pragma unroll
  for (int h = 0; h < 4; ++h) {
    float w = expf(ae[(size_t)e * 4 + h] - A);
    ae[(size_t)e * 4 + h] = w;                   // w > 0 -> raw bits are ordered
    atomicMax(&mseg[r * 4 + h], __float_as_uint(w));
  }
}

// ---------------- edge pass 3: e = exp(w - m); segment sum -------------------
__global__ __launch_bounds__(256) void k_edge3(const int* __restrict__ ei,
                                               float* __restrict__ ae,
                                               const unsigned* __restrict__ mseg,
                                               float* __restrict__ denom) {
  int e = blockIdx.x * 256 + threadIdx.x;
  if (e >= EE) return;
  int r = ei[e];
#pragma unroll
  for (int h = 0; h < 4; ++h) {
    float w  = ae[(size_t)e * 4 + h];
    float mf = __uint_as_float(mseg[r * 4 + h]);
    float ex = expf(w - mf);
    ae[(size_t)e * 4 + h] = ex;
    atomicAdd(&denom[r * 4 + h], ex);
  }
}

// ---------------- edge pass 4: head-averaged weighted scatter ----------------
__global__ __launch_bounds__(256) void k_edge4(const int* __restrict__ ei,
                                               const float* __restrict__ ae,
                                               const float* __restrict__ denom,
                                               const _Float16* __restrict__ xph,
                                               float* __restrict__ gat_acc) {
  size_t t = (size_t)blockIdx.x * 256 + threadIdx.x;   // t = e*128 + d
  if (t >= (size_t)EE * 128) return;
  int e = (int)(t >> 7), d = (int)(t & 127);
  int r = ei[e], c = ei[EE + e];
  float s = 0.f;
#pragma unroll
  for (int h = 0; h < 4; ++h) {
    float alpha = ae[(size_t)e * 4 + h] / denom[r * 4 + h];
    s += alpha * (float)xph[(size_t)c * HD + h * DD + d];
  }
  atomicAdd(&gat_acc[(size_t)r * DD + d], 0.25f * s);
}

// ---------------- gated residual: x_state = x + tanh(gate)*gat_mean ---------
__global__ __launch_bounds__(256) void k_resid_gate(const float* __restrict__ x,
                                                    const float* __restrict__ acc,
                                                    const float* __restrict__ gate,
                                                    float* __restrict__ xs) {
  int i = blockIdx.x * 256 + threadIdx.x;
  if (i >= NN * DD) return;
  xs[i] = x[i] + tanhf(gate[0]) * acc[i];
}

// ---------------- LayerNorm: one wave32 per row (4 elems / lane) -------------
__global__ __launch_bounds__(256) void k_ln(const float* __restrict__ X,
                                            const float* __restrict__ g,
                                            const float* __restrict__ b,
                                            _Float16* __restrict__ Y, int rows) {
  int wave = threadIdx.x >> 5, lane = threadIdx.x & 31;
  int row = blockIdx.x * 8 + wave;
  if (row >= rows) return;
  const float* xr = X + (size_t)row * 128;
  float4 v = ((const float4*)xr)[lane];
  float s = v.x + v.y + v.z + v.w;
  for (int o = 16; o > 0; o >>= 1) s += __shfl_xor(s, o, 32);
  float mu = s * (1.f / 128.f);
  float d0 = v.x - mu, d1 = v.y - mu, d2 = v.z - mu, d3 = v.w - mu;
  float q = d0 * d0 + d1 * d1 + d2 * d2 + d3 * d3;
  for (int o = 16; o > 0; o >>= 1) q += __shfl_xor(q, o, 32);
  float rs = rsqrtf(q * (1.f / 128.f) + 1.0e-5f);
  int c0 = lane * 4;
  _Float16* yr = Y + (size_t)row * 128 + c0;
  yr[0] = (_Float16)(d0 * rs * g[c0 + 0] + b[c0 + 0]);
  yr[1] = (_Float16)(d1 * rs * g[c0 + 1] + b[c0 + 1]);
  yr[2] = (_Float16)(d2 * rs * g[c0 + 2] + b[c0 + 2]);
  yr[3] = (_Float16)(d3 * rs * g[c0 + 3] + b[c0 + 3]);
}

// ---------------- K_sum[h,m] = sum_n Kp (reads contiguous KpT rows) ----------
__global__ __launch_bounds__(256) void k_ksum(const _Float16* __restrict__ KpT,
                                              float* __restrict__ Ksum) {
  __shared__ float sm[256];
  int row = blockIdx.x;                          // 0..511  (h*128 + m)
  float s = 0.f;
  for (int i = threadIdx.x; i < NPAD; i += 256)
    s += (float)KpT[(size_t)row * NPAD + i];
  sm[threadIdx.x] = s;
  __syncthreads();
  for (int o = 128; o > 0; o >>= 1) {
    if (threadIdx.x < o) sm[threadIdx.x] += sm[threadIdx.x + o];
    __syncthreads();
  }
  if (threadIdx.x == 0) Ksum[row] = sm[0];
}

// ---------------- KV[h][m][d] -> KV^T f16 [h][d][m] --------------------------
__global__ __launch_bounds__(256) void k_kvt(const float* __restrict__ KV,
                                             _Float16* __restrict__ KVhT) {
  int t = blockIdx.x * 256 + threadIdx.x;
  if (t >= 4 * 128 * 128) return;
  int h = t / 16384, rem = t % 16384, d = rem / 128, m = rem % 128;
  KVhT[(size_t)h * 16384 + d * 128 + m] = (_Float16)KV[(size_t)h * 16384 + m * 128 + d];
}

// ---------------- z[h*N+n] = 1 / (Qp . K_sum + 1e-6) -------------------------
__global__ __launch_bounds__(256) void k_z(const _Float16* __restrict__ Qph,
                                           const float* __restrict__ Ksum,
                                           float* __restrict__ z) {
  int r = blockIdx.x * 256 + threadIdx.x;       // r = h*N + n
  if (r >= NN * 4) return;
  int h = r / NN;
  const _Float16* q = Qph + (size_t)r * 128;
  float s = 0.f;
  for (int m = 0; m < 128; ++m) s += (float)q[m] * Ksum[h * 128 + m];
  z[r] = 1.0f / (s + 1.0e-6f);
}

// ===========================================================================
extern "C" void kernel_launch(void* const* d_in, const int* in_sizes, int n_in,
                              void* d_out, int out_size, void* d_ws, size_t ws_size,
                              hipStream_t stream) {
  (void)in_sizes; (void)n_in; (void)out_size; (void)ws_size;
  const float* x      = (const float*)d_in[0];
  const int*   ei     = (const int*)  d_in[1];
  const float* gat_w  = (const float*)d_in[2];
  const float* attn_l = (const float*)d_in[3];
  const float* attn_r = (const float*)d_in[4];
  const float* gate   = (const float*)d_in[5];
  const float* ln1_g  = (const float*)d_in[6];
  const float* ln1_b  = (const float*)d_in[7];
  const float* q_w    = (const float*)d_in[8];
  const float* k_w    = (const float*)d_in[9];
  const float* v_w    = (const float*)d_in[10];
  const float* out_w  = (const float*)d_in[11];
  const float* out_b  = (const float*)d_in[12];
  const float* proj   = (const float*)d_in[13];
  const float* ln2_g  = (const float*)d_in[14];
  const float* ln2_b  = (const float*)d_in[15];
  const float* w1     = (const float*)d_in[16];
  const float* b1     = (const float*)d_in[17];
  const float* w2     = (const float*)d_in[18];
  const float* b2     = (const float*)d_in[19];
  float* outp = (float*)d_out;

  // ---- workspace carve-up (256B aligned) ----
  size_t cur = 0;
  auto alloc = [&](size_t bytes) -> void* {
    size_t p = (cur + 255) & ~size_t(255);
    cur = p + bytes;
    return (void*)((char*)d_ws + p);
  };
  _Float16* gat_wh = (_Float16*)alloc((size_t)HD * DD * 2);
  _Float16* q_wh   = (_Float16*)alloc((size_t)HD * DD * 2);
  _Float16* k_wh   = (_Float16*)alloc((size_t)HD * DD * 2);
  _Float16* v_wh   = (_Float16*)alloc((size_t)HD * DD * 2);
  _Float16* out_wh = (_Float16*)alloc((size_t)DD * HD * 2);
  _Float16* projh  = (_Float16*)alloc((size_t)DD * DD * 2);
  _Float16* w1h    = (_Float16*)alloc((size_t)HD * DD * 2);
  _Float16* w2h    = (_Float16*)alloc((size_t)DD * HD * 2);

  float*    xstate = (float*)   alloc((size_t)NN * DD * 4);
  _Float16* buf16  = (_Float16*)alloc((size_t)NN * DD * 2);   // xh / yh / y2h
  _Float16* bufA   = (_Float16*)alloc((size_t)NN * HD * 2);   // xph -> Qh -> AOh
  _Float16* bufB   = (_Float16*)alloc((size_t)NN * HD * 2);   // Kh -> h1h
  _Float16* VhT    = (_Float16*)alloc((size_t)HD * NPAD * 2);
  _Float16* KpT    = (_Float16*)alloc((size_t)HD * NPAD * 2);
  _Float16* Qph    = (_Float16*)alloc((size_t)NN * HD * 2);   // head-major [H,N,128]
  float*    al     = (float*)   alloc((size_t)NN * 4 * 4);
  float*    ar     = (float*)   alloc((size_t)NN * 4 * 4);
  float*    ae     = (float*)   alloc((size_t)EE * 4 * 4);
  unsigned* mseg   = (unsigned*)alloc((size_t)NN * 4 * 4);
  float*    denom  = (float*)   alloc((size_t)NN * 4 * 4);
  float*    gacc   = (float*)   alloc((size_t)NN * DD * 4);
  unsigned* amax   = (unsigned*)alloc(256);
  float*    Ksum   = (float*)   alloc(512 * 4);
  float*    KV     = (float*)   alloc((size_t)4 * 128 * 128 * 4);
  _Float16* KVhT   = (_Float16*)alloc((size_t)4 * 128 * 128 * 2);
  float*    zbuf   = (float*)   alloc((size_t)NN * 4 * 4);

  const float dn = 1.0f / powf((float)DD, 0.25f);   // D ** -0.25

  // ---- weight conversion f32 -> f16 ----
  k_f16cvt<<<256, 256, 0, stream>>>(gat_wh, gat_w, HD * DD);
  k_f16cvt<<<256, 256, 0, stream>>>(q_wh,   q_w,   HD * DD);
  k_f16cvt<<<256, 256, 0, stream>>>(k_wh,   k_w,   HD * DD);
  k_f16cvt<<<256, 256, 0, stream>>>(v_wh,   v_w,   HD * DD);
  k_f16cvt<<<256, 256, 0, stream>>>(out_wh, out_w, DD * HD);
  k_f16cvt<<<256, 256, 0, stream>>>(projh,  proj,  DD * DD);
  k_f16cvt<<<256, 256, 0, stream>>>(w1h,    w1,    HD * DD);
  k_f16cvt<<<256, 256, 0, stream>>>(w2h,    w2,    DD * HD);

  // ---- accumulator init (every call: ws is not re-zeroed by harness) ----
  k_zero_u32<<<4096, 256, 0, stream>>>((unsigned*)gacc,  (size_t)NN * DD);
  k_zero_u32<<<64,   256, 0, stream>>>(mseg,             (size_t)NN * 4);
  k_zero_u32<<<64,   256, 0, stream>>>((unsigned*)denom, (size_t)NN * 4);
  k_zero_u32<<<64,   256, 0, stream>>>((unsigned*)KV,    (size_t)4 * 128 * 128);
  k_zero_u32<<<4096, 256, 0, stream>>>((unsigned*)VhT,   (size_t)HD * NPAD / 2);
  k_zero_u32<<<4096, 256, 0, stream>>>((unsigned*)KpT,   (size_t)HD * NPAD / 2);
  k_init_amax<<<1, 1, 0, stream>>>(amax);

  // =================== Stage 1: gated sparse GAT ===================
  k_f16cvt<<<4096, 256, 0, stream>>>(buf16, x, NN * DD);              // xh
  {
    EpiP P{}; P.oh = bufA; P.Cfull = HD;                              // xph [N,512]
    int cbl = HD / 64, tiles = (NN / 16) * cbl;
    k_gemm<M_STORE16><<<(tiles + 7) / 8, 256, 0, stream>>>(buf16, gat_wh, DD, cbl, tiles, P);
  }
  k_alar <<<(NN * 4 + 255) / 256, 256, 0, stream>>>(bufA, attn_l, attn_r, al, ar);
  k_edge1<<<(EE + 255) / 256, 256, 0, stream>>>(ei, al, ar, ae, amax);
  k_edge2<<<(EE + 255) / 256, 256, 0, stream>>>(ei, ae, amax, mseg);
  k_edge3<<<(EE + 255) / 256, 256, 0, stream>>>(ei, ae, mseg, denom);
  k_edge4<<<(unsigned)(((size_t)EE * 128 + 255) / 256), 256, 0, stream>>>(ei, ae, denom, bufA, gacc);
  k_resid_gate<<<(NN * DD + 255) / 256, 256, 0, stream>>>(x, gacc, gate, xstate);

  // =================== Stage 2: performer attention ===================
  k_ln<<<(NN + 7) / 8, 256, 0, stream>>>(xstate, ln1_g, ln1_b, buf16, NN);  // yh
  {
    int cbl = HD / 64, tiles = (NN / 16) * cbl;
    EpiP Pq{}; Pq.oh = bufA; Pq.Cfull = HD;                           // Qh
    k_gemm<M_STORE16><<<(tiles + 7) / 8, 256, 0, stream>>>(buf16, q_wh, DD, cbl, tiles, Pq);
    EpiP Pk{}; Pk.oh = bufB; Pk.Cfull = HD;                           // Kh
    k_gemm<M_STORE16><<<(tiles + 7) / 8, 256, 0, stream>>>(buf16, k_wh, DD, cbl, tiles, Pk);
    EpiP Pv{}; Pv.oh = VhT; Pv.ldT = NPAD;                            // V^T staged
    k_gemm<M_STOREVT><<<(tiles + 7) / 8, 256, 0, stream>>>(buf16, v_wh, DD, cbl, tiles, Pv);
  }
  {
    int cbl = DD / 64, tiles = ((NN * 4) / 16) * cbl;
    EpiP Pq{}; Pq.oh = Qph; Pq.Nnodes = NN; Pq.dn = dn;               // Qp head-major
    k_gemm<M_QP><<<(tiles + 7) / 8, 256, 0, stream>>>(bufA, projh, DD, cbl, tiles, Pq);
    EpiP Pk{}; Pk.oh = KpT; Pk.ldT = NPAD; Pk.dn = dn;                // Kp^T
    k_gemm<M_KP><<<(tiles + 7) / 8, 256, 0, stream>>>(bufB, projh, DD, cbl, tiles, Pk);
  }
  k_ksum<<<512, 256, 0, stream>>>(KpT, Ksum);
  k_kv  <<<(4 * 4 * 4 * 64 + 7) / 8, 256, 0, stream>>>(KpT, VhT, KV, 64, NPAD / 32);
  k_kvt <<<(4 * 128 * 128 + 255) / 256, 256, 0, stream>>>(KV, KVhT);
  k_z   <<<(NN * 4 + 255) / 256, 256, 0, stream>>>(Qph, Ksum, zbuf);
  {
    int cbl = DD / 64, tiles = ((NN * 4) / 16) * cbl;
    EpiP P{}; P.oh = bufA; P.zbuf = zbuf; P.Nnodes = NN;              // AOh [N,512]
    k_gemm<M_OUTEIN><<<(tiles + 7) / 8, 256, 0, stream>>>(Qph, KVhT, DD, cbl, tiles, P);
  }
  {
    int cbl = DD / 64, tiles = (NN / 16) * cbl;
    EpiP P{}; P.of = xstate; P.bias = out_b; P.Cfull = DD;            // x_state += attn
    k_gemm<M_RESADD><<<(tiles + 7) / 8, 256, 0, stream>>>(bufA, out_wh, HD, cbl, tiles, P);
  }

  // =================== Stage 3: FFN ===================
  k_ln<<<(NN + 7) / 8, 256, 0, stream>>>(xstate, ln2_g, ln2_b, buf16, NN);  // y2h
  {
    int cbl = HD / 64, tiles = (NN / 16) * cbl;
    EpiP P{}; P.oh = bufB; P.bias = b1; P.Cfull = HD;                 // h1h = gelu(...)
    k_gemm<M_GELU16><<<(tiles + 7) / 8, 256, 0, stream>>>(buf16, w1h, DD, cbl, tiles, P);
  }
  {
    int cbl = DD / 64, tiles = (NN / 16) * cbl;
    EpiP P{}; P.of = outp; P.resid = xstate; P.bias = b2; P.Cfull = DD; // d_out
    k_gemm<M_FINAL><<<(tiles + 7) / 8, 256, 0, stream>>>(bufB, w2h, HD, cbl, tiles, P);
  }
}
